// ResidualMamba_17162689315093
// MI455X (gfx1250) — matrix operand order repaired and checked by reading, not compiled, not run on hardware
//
#include <hip/hip_runtime.h>
#include <hip/hip_bf16.h>
#include <stddef.h>

#define D_MODEL 1024
#define D_STATE 16
#define D_CONV  4
#define D_INNER 2048
#define DT_RANK 64
#define BATCH   2
#define SEQ     2048
#define NTOK    (BATCH * SEQ)             // 4096 tokens
#define XDBL_N  (DT_RANK + 2 * D_STATE)   // 96

typedef __attribute__((ext_vector_type(16))) _Float16 v16h;
typedef __attribute__((ext_vector_type(8)))  float    v8f;

__device__ __forceinline__ float silu_f(float v) { return v / (1.0f + __expf(-v)); }

// ---------------------------------------------------------------- CDNA5 async-to-LDS support
#if defined(__HIP_DEVICE_COMPILE__) && __has_builtin(__builtin_amdgcn_global_load_async_to_lds_b128)
#define HAVE_ASYNC_LDS 1
typedef int v4i32 __attribute__((vector_size(16)));
typedef __attribute__((address_space(1))) v4i32 glob_v4i;   // pointee in AS1 (global)
typedef __attribute__((address_space(3))) v4i32 lds_v4i;    // pointee in AS3 (LDS)
#else
#define HAVE_ASYNC_LDS 0
#endif

__device__ __forceinline__ void wait_async0() {
#if HAVE_ASYNC_LDS
#if __has_builtin(__builtin_amdgcn_s_wait_asynccnt)
    __builtin_amdgcn_s_wait_asynccnt(0);
#else
    asm volatile("s_wait_asynccnt 0" ::: "memory");
#endif
#endif
}

// ---------------------------------------------------------------- convert
__global__ void k_f32_to_f16(const float* __restrict__ s, _Float16* __restrict__ d, int n) {
    int i = blockIdx.x * 256 + threadIdx.x;
    if (i < n) d[i] = (_Float16)s[i];
}

// ---------------------------------------------------------------- layernorm -> f16
__global__ __launch_bounds__(256) void k_layernorm(
    const float* __restrict__ x, const float* __restrict__ w,
    const float* __restrict__ b, _Float16* __restrict__ xn16) {
    __shared__ float red[256];
    __shared__ float s_mu, s_rs;
    int tok = blockIdx.x, t = threadIdx.x;
    float4 v = ((const float4*)x)[tok * (D_MODEL / 4) + t];
    red[t] = v.x + v.y + v.z + v.w;
    __syncthreads();
    for (int st = 128; st > 0; st >>= 1) { if (t < st) red[t] += red[t + st]; __syncthreads(); }
    if (t == 0) s_mu = red[0] * (1.0f / D_MODEL);
    __syncthreads();
    float mu = s_mu;
    float d0 = v.x - mu, d1 = v.y - mu, d2 = v.z - mu, d3 = v.w - mu;
    red[t] = d0 * d0 + d1 * d1 + d2 * d2 + d3 * d3;
    __syncthreads();
    for (int st = 128; st > 0; st >>= 1) { if (t < st) red[t] += red[t + st]; __syncthreads(); }
    if (t == 0) s_rs = rsqrtf(red[0] * (1.0f / D_MODEL) + 1e-5f);
    __syncthreads();
    float rs = s_rs;
    float4 w4 = ((const float4*)w)[t];
    float4 b4 = ((const float4*)b)[t];
    size_t o = (size_t)tok * D_MODEL + 4 * t;
    xn16[o + 0] = (_Float16)(d0 * rs * w4.x + b4.x);
    xn16[o + 1] = (_Float16)(d1 * rs * w4.y + b4.y);
    xn16[o + 2] = (_Float16)(d2 * rs * w4.z + b4.z);
    xn16[o + 3] = (_Float16)(d3 * rs * w4.w + b4.w);
}

// ---------------------------------------------------------------- WMMA GEMM (double-buffered)
// C[M,N] (f32) = A[M,K] (f16, row-major) * B[K,N] (f16, row-major)
// 256 threads = 8 waves; block tile 128x64, K-step 32, 2-deep LDS pipeline.
#define BM 128
#define BN 64
#define BK 32

// A tile 128x32: 512 chunks of 8 halves (16B); thread does 2. Async DMA into LDS when available.
__device__ __forceinline__ void copyA(const _Float16* __restrict__ A, _Float16* As,
                                      int t, int m0, int k0, int K) {
    #pragma unroll
    for (int i = 0; i < 2; ++i) {
        int c   = t + i * 256;
        int row = c >> 2;
        int col = (c & 3) * 8;
        const _Float16* src = &A[(size_t)(m0 + row) * K + k0 + col];
        _Float16*       dst = &As[row * BK + col];
#if HAVE_ASYNC_LDS
        __builtin_amdgcn_global_load_async_to_lds_b128(
            (glob_v4i*)src, (lds_v4i*)dst, 0, 0);
#else
        *(float4*)dst = *(const float4*)src;
#endif
    }
}

// B tile 32x64: each thread one 8-half chunk along N. Branchless edge clamp.
__device__ __forceinline__ void loadB(const _Float16* __restrict__ B, int t, int n0,
                                      int k0, int N, _Float16 tmp[8]) {
    int krow = t >> 3;
    int nc   = (t & 7) * 8;
    const _Float16* Brow = &B[(size_t)(k0 + krow) * N];
    if (n0 + nc + 7 < N) {
        *(float4*)tmp = *(const float4*)(&Brow[n0 + nc]);
    } else {
        #pragma unroll
        for (int j = 0; j < 8; ++j) {
            int col = n0 + nc + j;
            int cc  = col < N ? col : N - 1;        // clamp address: load always executes
            _Float16 v = Brow[cc];
            tmp[j] = col < N ? v : (_Float16)0.0f;  // v_cndmask, no exec branching
        }
    }
}

__device__ __forceinline__ void storeB(_Float16* Bs, int t, const _Float16 tmp[8]) {
    int krow = t >> 3;
    int nc   = (t & 7) * 8;
    #pragma unroll
    for (int j = 0; j < 8; ++j) Bs[(nc + j) * BK + krow] = tmp[j];   // transposed [n][k]
}

__global__ __launch_bounds__(256) void k_gemm_f16(
    const _Float16* __restrict__ A, const _Float16* __restrict__ B,
    float* __restrict__ C, int M, int N, int K) {
    __shared__ __align__(32) _Float16 As[2][BM * BK];
    __shared__ __align__(32) _Float16 Bs[2][BN * BK];
    int m0 = blockIdx.y * BM;
    int n0 = blockIdx.x * BN;
    int t = threadIdx.x;
    int wave = t >> 5;
    int lane = t & 31;

    v8f acc[4] = {};
    _Float16 btmp[8];

    // prologue: fill pipeline stage 0
    copyA(A, As[0], t, m0, 0, K);
    loadB(B, t, n0, 0, N, btmp);
    storeB(Bs[0], t, btmp);

    int cur = 0;
    for (int k0 = 0; k0 < K; k0 += BK) {
        wait_async0();            // my async writes into As[cur] complete
        __syncthreads();          // everyone's tile-k data visible; prev buffer free
        int nxt  = cur ^ 1;
        bool more = (k0 + BK) < K;
        if (more) {
            loadB(B, t, n0, k0 + BK, N, btmp);   // global loads in flight over the WMMAs
            copyA(A, As[nxt], t, m0, k0 + BK, K);
        }

        // A fragment (16x32 f16): lane<16 -> K {0..7,16..23}; lane>=16 -> K {8..15,24..31}
        int arow = wave * 16 + (lane & 15);
        int kl   = (lane >> 4) * 8;
        const _Float16* Ab = &As[cur][arow * BK];
        v16h af;
        *((float4*)&af)       = *(const float4*)(Ab + kl);
        *(((float4*)&af) + 1) = *(const float4*)(Ab + kl + 16);

        // all 4 B fragments, then 4 back-to-back WMMAs
        v16h bf[4];
        #pragma unroll
        for (int nt = 0; nt < 4; ++nt) {
            const _Float16* Bb = &Bs[cur][(nt * 16 + (lane & 15)) * BK + (lane >> 4) * 8];
            *((float4*)&bf[nt])       = *(const float4*)(Bb);
            *(((float4*)&bf[nt]) + 1) = *(const float4*)(Bb + 16);
        }
        #pragma unroll
        for (int nt = 0; nt < 4; ++nt)
            acc[nt] = __builtin_amdgcn_wmma_f32_16x16x32_f16(
                false, af, false, bf[nt], (short)0, acc[nt], false, false);

        if (more) storeB(Bs[nxt], t, btmp);      // B wait sinks below the WMMAs
        cur = nxt;
    }

    // C/D layout: VGPR r -> M = r (+8 for lanes 16..31), N = lane&15
    int mrow  = m0 + wave * 16 + (lane >> 4) * 8;
    int ncol0 = n0 + (lane & 15);
    #pragma unroll
    for (int nt = 0; nt < 4; ++nt) {
        int ncol = ncol0 + nt * 16;
        if (ncol < N) {
            #pragma unroll
            for (int r = 0; r < 8; ++r)
                C[(size_t)(mrow + r) * N + ncol] = acc[nt][r];
        }
    }
}

// ---------------------------------------------------------------- depthwise causal conv + SiLU
__global__ void k_conv_silu(const float* __restrict__ xz, const float* __restrict__ cw,
                            const float* __restrict__ cb, float* __restrict__ u32,
                            _Float16* __restrict__ u16) {
    int idx = blockIdx.x * 256 + threadIdx.x;     // over NTOK*D_INNER
    if (idx >= NTOK * D_INNER) return;
    int c  = idx & (D_INNER - 1);
    int bl = idx >> 11;
    int l  = bl & (SEQ - 1);
    float acc = cb[c];
    #pragma unroll
    for (int j = 0; j < D_CONV; ++j) {
        int dl = l - (D_CONV - 1) + j;
        if (dl >= 0)
            acc += cw[c * D_CONV + j] *
                   xz[(size_t)(bl - (D_CONV - 1) + j) * (2 * D_INNER) + c];
    }
    float v = silu_f(acc);
    u32[idx] = v;
    u16[idx] = (_Float16)v;
}

// ---------------------------------------------------------------- dt slice -> f16
__global__ void k_dtsplit(const float* __restrict__ xdbl, _Float16* __restrict__ dt16) {
    int idx = blockIdx.x * 256 + threadIdx.x;     // over NTOK*DT_RANK
    if (idx >= NTOK * DT_RANK) return;
    int r = idx & (DT_RANK - 1);
    int row = idx >> 6;
    dt16[idx] = (_Float16)xdbl[(size_t)row * XDBL_N + r];
}

// ---------------------------------------------------------------- softplus(delta_pre + b_dt)
__global__ void k_softplus_bias(float* __restrict__ delta, const float* __restrict__ bdt) {
    int idx = blockIdx.x * 256 + threadIdx.x;     // over NTOK*D_INNER
    if (idx >= NTOK * D_INNER) return;
    float v = delta[idx] + bdt[idx & (D_INNER - 1)];
    delta[idx] = (v > 20.0f) ? v : log1pf(__expf(v));
}

// ---------------------------------------------------------------- selective scan
// One lane per (batch, channel); h[16] and A-row in registers; L sequential.
__global__ __launch_bounds__(256) void k_scan(
    const float* __restrict__ delta, const float* __restrict__ u,
    const float* __restrict__ xdbl, const float* __restrict__ A_log,
    float* __restrict__ ys) {
    int ch = blockIdx.x * 256 + threadIdx.x;      // 0 .. BATCH*D_INNER-1
    if (ch >= BATCH * D_INNER) return;
    int b = ch >> 11;                              // / D_INNER
    int c = ch & (D_INNER - 1);
    float Ar[D_STATE], h[D_STATE];
    #pragma unroll
    for (int s = 0; s < D_STATE; ++s) {
        Ar[s] = -__expf(A_log[c * D_STATE + s]);
        h[s]  = 0.0f;
    }
    for (int l = 0; l < SEQ; ++l) {
        size_t row = (size_t)b * SEQ + l;
        float d  = delta[row * D_INNER + c];
        float uu = u[row * D_INNER + c];
        float du = d * uu;
        const float* Bt = &xdbl[row * XDBL_N + DT_RANK];   // uniform -> scalar loads
        const float* Ct = Bt + D_STATE;
        float y = 0.0f;
        #pragma unroll
        for (int s = 0; s < D_STATE; ++s) {
            h[s] = __expf(d * Ar[s]) * h[s] + du * Bt[s];
            y += h[s] * Ct[s];
        }
        ys[row * D_INNER + c] = y;
    }
}

// ---------------------------------------------------------------- gate: (ys + u*D)*silu(z) -> f16
__global__ void k_gate(const float* __restrict__ ys, const float* __restrict__ u32,
                       const float* __restrict__ xz, const float* __restrict__ Dskip,
                       _Float16* __restrict__ y16) {
    int idx = blockIdx.x * 256 + threadIdx.x;     // over NTOK*D_INNER
    if (idx >= NTOK * D_INNER) return;
    int c  = idx & (D_INNER - 1);
    int bl = idx >> 11;
    float z = xz[(size_t)bl * (2 * D_INNER) + D_INNER + c];
    float y = (ys[idx] + u32[idx] * Dskip[c]) * silu_f(z);
    y16[idx] = (_Float16)y;
}

// ---------------------------------------------------------------- erf-GELU + residual
__global__ void k_out(const float* __restrict__ outp, const float* __restrict__ x,
                      float* __restrict__ out) {
    int idx = blockIdx.x * 256 + threadIdx.x;     // over NTOK*D_MODEL
    if (idx >= NTOK * D_MODEL) return;
    float v = outp[idx];
    out[idx] = 0.5f * v * (1.0f + erff(v * 0.70710678118654752f)) + x[idx];
}

// ================================================================ launch
extern "C" void kernel_launch(void* const* d_in, const int* in_sizes, int n_in,
                              void* d_out, int out_size, void* d_ws, size_t ws_size,
                              hipStream_t stream) {
    const float* x      = (const float*)d_in[0];
    const float* ln_w   = (const float*)d_in[1];
    const float* ln_b   = (const float*)d_in[2];
    const float* W_in   = (const float*)d_in[3];
    const float* conv_w = (const float*)d_in[4];
    const float* conv_b = (const float*)d_in[5];
    const float* W_x    = (const float*)d_in[6];
    const float* W_dt   = (const float*)d_in[7];
    const float* b_dt   = (const float*)d_in[8];
    const float* A_log  = (const float*)d_in[9];
    const float* D_skip = (const float*)d_in[10];
    const float* W_out  = (const float*)d_in[11];
    float* out = (float*)d_out;

    char* ws = (char*)d_ws;
    size_t off = 0;
    auto alloc = [&](size_t bytes) -> void* {
        void* p = ws + off;
        off += bytes;
        off = (off + 255) & ~(size_t)255;
        return p;
    };
    _Float16* W_in16  = (_Float16*)alloc((size_t)D_MODEL * 2 * D_INNER * 2);
    _Float16* W_x16   = (_Float16*)alloc((size_t)D_INNER * XDBL_N * 2);
    _Float16* W_dt16  = (_Float16*)alloc((size_t)DT_RANK * D_INNER * 2);
    _Float16* W_out16 = (_Float16*)alloc((size_t)D_INNER * D_MODEL * 2);
    _Float16* xn16    = (_Float16*)alloc((size_t)NTOK * D_MODEL * 2);
    float*    xz      = (float*)   alloc((size_t)NTOK * 2 * D_INNER * 4);
    _Float16* u16     = (_Float16*)alloc((size_t)NTOK * D_INNER * 2);   // reused as y16
    float*    u32     = (float*)   alloc((size_t)NTOK * D_INNER * 4);
    float*    xdbl    = (float*)   alloc((size_t)NTOK * XDBL_N * 4);
    _Float16* dt16    = (_Float16*)alloc((size_t)NTOK * DT_RANK * 2);
    float*    delta   = (float*)   alloc((size_t)NTOK * D_INNER * 4);   // reused as outp
    float*    ys      = (float*)   alloc((size_t)NTOK * D_INNER * 4);
    _Float16* y16  = u16;     // u16 dead after GEMM2
    float*    outp = delta;   // delta dead after scan

    auto cdiv = [](int a, int b) { return (a + b - 1) / b; };

    // 1. weights -> f16
    k_f32_to_f16<<<cdiv(D_MODEL * 2 * D_INNER, 256), 256, 0, stream>>>(W_in,  W_in16,  D_MODEL * 2 * D_INNER);
    k_f32_to_f16<<<cdiv(D_INNER * XDBL_N,     256), 256, 0, stream>>>(W_x,   W_x16,   D_INNER * XDBL_N);
    k_f32_to_f16<<<cdiv(DT_RANK * D_INNER,    256), 256, 0, stream>>>(W_dt,  W_dt16,  DT_RANK * D_INNER);
    k_f32_to_f16<<<cdiv(D_INNER * D_MODEL,    256), 256, 0, stream>>>(W_out, W_out16, D_INNER * D_MODEL);

    // 2. LayerNorm
    k_layernorm<<<NTOK, 256, 0, stream>>>(x, ln_w, ln_b, xn16);

    // 3. xz = xn @ W_in   [4096 x 4096]
    k_gemm_f16<<<dim3(cdiv(2 * D_INNER, BN), NTOK / BM), 256, 0, stream>>>(
        xn16, W_in16, xz, NTOK, 2 * D_INNER, D_MODEL);

    // 4. depthwise conv + SiLU -> u
    k_conv_silu<<<cdiv(NTOK * D_INNER, 256), 256, 0, stream>>>(xz, conv_w, conv_b, u32, u16);

    // 5. x_dbl = u @ W_x   [4096 x 96]
    k_gemm_f16<<<dim3(cdiv(XDBL_N, BN), NTOK / BM), 256, 0, stream>>>(
        u16, W_x16, xdbl, NTOK, XDBL_N, D_INNER);

    // 6. dt slice -> f16
    k_dtsplit<<<cdiv(NTOK * DT_RANK, 256), 256, 0, stream>>>(xdbl, dt16);

    // 7. delta_pre = dt @ W_dt   [4096 x 2048]
    k_gemm_f16<<<dim3(cdiv(D_INNER, BN), NTOK / BM), 256, 0, stream>>>(
        dt16, W_dt16, delta, NTOK, D_INNER, DT_RANK);

    // 8. delta = softplus(delta_pre + b_dt)
    k_softplus_bias<<<cdiv(NTOK * D_INNER, 256), 256, 0, stream>>>(delta, b_dt);

    // 9. selective scan
    k_scan<<<cdiv(BATCH * D_INNER, 256), 256, 0, stream>>>(delta, u32, xdbl, A_log, ys);

    // 10. gate -> y16
    k_gate<<<cdiv(NTOK * D_INNER, 256), 256, 0, stream>>>(ys, u32, xz, D_skip, y16);

    // 11. outp = y @ W_out   [4096 x 1024]
    k_gemm_f16<<<dim3(cdiv(D_MODEL, BN), NTOK / BM), 256, 0, stream>>>(
        y16, W_out16, outp, NTOK, D_MODEL, D_INNER);

    // 12. GELU + residual
    k_out<<<cdiv(NTOK * D_MODEL, 256), 256, 0, stream>>>(outp, x, out);
}
